// PIDMultiHeadAttention_56349970924064
// MI455X (gfx1250) — matrix operand order repaired and checked by reading, not compiled, not run on hardware
//
#include <hip/hip_runtime.h>

// ---------------------------------------------------------------------------
// Problem constants (from the reference)
// ---------------------------------------------------------------------------
#define BB     2
#define TT     2048
#define DMODEL 1024
#define NHEADS 16
#define DHEAD  64
#define MTOK   (BB * TT)          // 4096 rows of the (B*T, D) matrices

typedef __attribute__((ext_vector_type(16))) __bf16 v16bf;
typedef __attribute__((ext_vector_type(8)))  __bf16 v8bf;
typedef __attribute__((ext_vector_type(8)))  float  v8f;

// ---------------------------------------------------------------------------
// WMMA helper: D = A(16x32 bf16) * B(32x16 bf16) + C(16x16 f32)
// ---------------------------------------------------------------------------
__device__ __forceinline__ v8f wmma_bf16(v16bf a, v16bf b, v8f c) {
  return __builtin_amdgcn_wmma_f32_16x16x32_bf16(
      /*neg_a=*/false, a, /*neg_b=*/false, b,
      /*c_mod=*/(short)0, c, /*reuse_a=*/false, /*reuse_b=*/false);
}

// A-matrix fragment, 16x32 bf16, contiguous K order (matches load_b_frag):
//   lanes 0-15 : M=lane,     elems[0..15]=K k0+0..15
//   lanes16-31 : M=lane-16,  elems[0..15]=K k0+16..31
// Source is row-major with leading dimension ld (elements).
__device__ __forceinline__ v16bf load_a_frag(const __bf16* base, int ld,
                                             int row0, int k0, int lane) {
  int m  = row0 + (lane & 15);
  int kb = k0 + ((lane & 16) ? 16 : 0);
  const __bf16* p = base + (size_t)m * ld + kb;
  v8bf lo = *(const v8bf*)(p);
  v8bf hi = *(const v8bf*)(p + 8);
  v16bf r;
#pragma unroll
  for (int i = 0; i < 8; ++i) { r[i] = lo[i]; r[i + 8] = hi[i]; }
  return r;
}

// B-matrix fragment, 32x16 (KxN) bf16, contiguous K order (matches load_a_frag):
//   lanes 0-15 : N=n0+lane,     elems[0..15]=K k0+0..15
//   lanes16-31 : N=n0+lane-16,  elems[0..15]=K k0+16..31
// Source is "N-row-major" (each N row contiguous along K), i.e. W (NxK) row-major.
__device__ __forceinline__ v16bf load_b_frag(const __bf16* base, int ld,
                                             int n0, int k0, int lane) {
  int n  = n0 + (lane & 15);
  int kb = k0 + ((lane & 16) ? 16 : 0);
  const __bf16* p = base + (size_t)n * ld + kb;
  v8bf lo = *(const v8bf*)(p);
  v8bf hi = *(const v8bf*)(p + 8);
  v16bf r;
#pragma unroll
  for (int i = 0; i < 8; ++i) { r[i] = lo[i]; r[i + 8] = hi[i]; }
  return r;
}

// ---------------------------------------------------------------------------
// Kernel 1: convert the 10 f32 weight matrices (DxD each) to bf16
// ---------------------------------------------------------------------------
struct WPtrs { const float* p[10]; };

__global__ __launch_bounds__(256) void convert_weights_kernel(WPtrs wp, __bf16* out) {
  size_t gid = (size_t)blockIdx.x * 256 + threadIdx.x;   // 10 * 2^20 total
  int    mat = (int)(gid >> 20);
  size_t off = gid & 0xFFFFFull;
  off |= (gid & 0x100000ull) ? 0 : 0;                     // (no-op, keep simple)
  off = gid & ((1ull << 20) - 1);
  out[gid] = (__bf16)wp.p[mat][off];
}

// ---------------------------------------------------------------------------
// Kernel 2: gate softmax  g[p][m][c] = softmax_c(x[m,:] . Wg_p[c,:] + bg_p[c])
// One wave per (p, token).  128 threads = 4 waves per block.
// ---------------------------------------------------------------------------
__global__ __launch_bounds__(128) void pid_gates_kernel(
    const float* __restrict__ x,
    const float* __restrict__ qWg, const float* __restrict__ qbg,
    const float* __restrict__ kWg, const float* __restrict__ kbg,
    const float* __restrict__ vWg, const float* __restrict__ vbg,
    float* __restrict__ g) {
  int lane = threadIdx.x & 31;
  int wv   = blockIdx.x * 4 + (threadIdx.x >> 5);   // 0 .. 3*MTOK-1
  int p    = wv >> 12;                               // / 4096
  int m    = wv & (MTOK - 1);
  const float* Wg = (p == 0) ? qWg : (p == 1) ? kWg : vWg;
  const float* bg = (p == 0) ? qbg : (p == 1) ? kbg : vbg;
  const float* xr = x + (size_t)m * DMODEL;
  float s0 = 0.f, s1 = 0.f, s2 = 0.f;
  for (int i = lane; i < DMODEL; i += 32) {
    float xv = xr[i];
    s0 += xv * Wg[i];
    s1 += xv * Wg[DMODEL + i];
    s2 += xv * Wg[2 * DMODEL + i];
  }
#pragma unroll
  for (int off = 1; off < 32; off <<= 1) {
    s0 += __shfl_xor(s0, off, 32);
    s1 += __shfl_xor(s1, off, 32);
    s2 += __shfl_xor(s2, off, 32);
  }
  s0 += bg[0]; s1 += bg[1]; s2 += bg[2];
  float mx = fmaxf(s0, fmaxf(s1, s2));
  float e0 = __expf(s0 - mx), e1 = __expf(s1 - mx), e2 = __expf(s2 - mx);
  float inv = 1.0f / (e0 + e1 + e2);
  if (lane == 0) {
    float* gp = g + ((size_t)p * MTOK + m) * 3;
    gp[0] = e0 * inv; gp[1] = e1 * inv; gp[2] = e2 * inv;
  }
}

// ---------------------------------------------------------------------------
// Kernel 3: sequential scan along T per (b,d) channel.
// integral = cumsum/denominator, derivative = diff.  Emits bf16 copies.
// Coalesced: consecutive threads own consecutive d.
// ---------------------------------------------------------------------------
__global__ __launch_bounds__(256) void pid_scan_kernel(
    const float* __restrict__ x,
    __bf16* __restrict__ xbf, __bf16* __restrict__ intbf,
    __bf16* __restrict__ derbf) {
  int ch = blockIdx.x * 256 + threadIdx.x;   // 0 .. BB*DMODEL-1
  int b  = ch >> 10;
  int d  = ch & (DMODEL - 1);
  size_t base = (size_t)b * TT * DMODEL + d;
  float run = 0.f, prev = 0.f;
  for (int t = 0; t < TT; ++t) {
    size_t o  = base + (size_t)t * DMODEL;
    float  xv = x[o];
    run += xv;
    float integ = run / (float)(t + 1);
    float der   = (t == 0) ? 0.f : (xv - prev);
    prev = xv;
    xbf[o]   = (__bf16)xv;
    intbf[o] = (__bf16)integ;
    derbf[o] = (__bf16)der;
  }
}

// ---------------------------------------------------------------------------
// Kernel 4: fused PID projection GEMM.
//   out[m,n] = sum_c g[m,c] * (A_c @ W_c^T)[m,n]
// Block: 256 thr = 8 waves (4 along M x 2 along N); wave tile 32x32;
// block tile 128x64.  Three f32 accumulators per 16x16 tile, gate applied
// per-row in the epilogue.  Output scattered into head-major layouts:
//   mode 0/1 (Q,K):  out[(b*H+h)*T + t][dh]        (row-major, ld=DHEAD)
//   mode 2   (V):    Vt[(b*H+h)*DHEAD + dcol][t]   (transposed for P@V)
// ---------------------------------------------------------------------------
__global__ __launch_bounds__(256) void pid_gemm_kernel(
    const __bf16* __restrict__ A0, const __bf16* __restrict__ A1,
    const __bf16* __restrict__ A2,
    const __bf16* __restrict__ W0, const __bf16* __restrict__ W1,
    const __bf16* __restrict__ W2,
    const float* __restrict__ g,      // (MTOK, 3)
    __bf16* __restrict__ out, int mode) {
  int lane = threadIdx.x & 31;
  int wid  = threadIdx.x >> 5;
  int wm   = wid & 3, wn = wid >> 2;
  int m0   = blockIdx.x * 128 + wm * 32;
  int n0   = blockIdx.y * 64 + wn * 32;

  v8f acc[3][2][2] = {};
  const __bf16* As[3] = {A0, A1, A2};
  const __bf16* Ws[3] = {W0, W1, W2};

  for (int k0 = 0; k0 < DMODEL; k0 += 32) {
#pragma unroll
    for (int c = 0; c < 3; ++c) {
      v16bf a0 = load_a_frag(As[c], DMODEL, m0,      k0, lane);
      v16bf a1 = load_a_frag(As[c], DMODEL, m0 + 16, k0, lane);
      v16bf b0 = load_b_frag(Ws[c], DMODEL, n0,      k0, lane);
      v16bf b1 = load_b_frag(Ws[c], DMODEL, n0 + 16, k0, lane);
      acc[c][0][0] = wmma_bf16(a0, b0, acc[c][0][0]);
      acc[c][0][1] = wmma_bf16(a0, b1, acc[c][0][1]);
      acc[c][1][0] = wmma_bf16(a1, b0, acc[c][1][0]);
      acc[c][1][1] = wmma_bf16(a1, b1, acc[c][1][1]);
    }
  }

  int half  = lane >> 4;
  int nlane = lane & 15;
#pragma unroll
  for (int mi = 0; mi < 2; ++mi) {
#pragma unroll
    for (int ni = 0; ni < 2; ++ni) {
#pragma unroll
      for (int r = 0; r < 8; ++r) {
        int m = m0 + mi * 16 + r + 8 * half;
        int n = n0 + ni * 16 + nlane;
        const float* gr = g + (size_t)m * 3;
        float v = gr[0] * acc[0][mi][ni][r] +
                  gr[1] * acc[1][mi][ni][r] +
                  gr[2] * acc[2][mi][ni][r];
        int b = m >> 11, t = m & (TT - 1);
        int h = n >> 6,  dc = n & (DHEAD - 1);
        size_t addr;
        if (mode == 2)  // V transposed per head: (b*H+h, dcol, t)
          addr = ((size_t)(b * NHEADS + h) * DHEAD + dc) * TT + t;
        else            // Q / K: (b*H+h, t, dcol)
          addr = ((size_t)(b * NHEADS + h) * TT + t) * DHEAD + dc;
        out[addr] = (__bf16)v;
      }
    }
  }
}

// ---------------------------------------------------------------------------
// Kernel 5: causal flash attention, one wave per (b*H+h, 16-query tile).
// S = Q(16x64) @ K^T via 4 WMMAs per 32-key tile; online softmax with
// 16-lane shfl_xor row reductions (matches C-layout lane grouping);
// P restaged through LDS into A-fragment form; P@V via 4 WMMAs (dh=64).
// ---------------------------------------------------------------------------
__global__ __launch_bounds__(32) void flash_attn_kernel(
    const __bf16* __restrict__ Q, const __bf16* __restrict__ K,
    const __bf16* __restrict__ Vt, __bf16* __restrict__ O) {
  __shared__ __align__(16) __bf16 sP[16 * 32];

  int lane  = threadIdx.x & 31;
  int bh    = blockIdx.x;                 // b*NHEADS + h
  int qt0   = blockIdx.y * 16;            // query tile start (within head)
  int half  = lane >> 4;
  int nlane = lane & 15;

  const __bf16* Qh = Q  + (size_t)bh * TT * DHEAD;
  const __bf16* Kh = K  + (size_t)bh * TT * DHEAD;
  const __bf16* Vh = Vt + (size_t)bh * DHEAD * TT;

  v16bf aq0 = load_a_frag(Qh, DHEAD, qt0, 0,  lane);
  v16bf aq1 = load_a_frag(Qh, DHEAD, qt0, 32, lane);

  v8f   oacc[4] = {};
  float mrow[8], lrow[8];
#pragma unroll
  for (int r = 0; r < 8; ++r) { mrow[r] = -3.0e38f; lrow[r] = 0.f; }

  const float scale = 0.125f;             // 1/sqrt(64)

  for (int kt0 = 0; kt0 < qt0 + 16; kt0 += 32) {
    // ---- scores: two 16x16 C tiles covering keys [kt0, kt0+32) ----
    v8f s0 = {}, s1 = {};
    {
      v16bf b00 = load_b_frag(Kh, DHEAD, kt0,      0,  lane);
      v16bf b01 = load_b_frag(Kh, DHEAD, kt0,      32, lane);
      s0 = wmma_bf16(aq0, b00, s0);
      s0 = wmma_bf16(aq1, b01, s0);
      v16bf b10 = load_b_frag(Kh, DHEAD, kt0 + 16, 0,  lane);
      v16bf b11 = load_b_frag(Kh, DHEAD, kt0 + 16, 32, lane);
      s1 = wmma_bf16(aq0, b10, s1);
      s1 = wmma_bf16(aq1, b11, s1);
    }

    int key0 = kt0 + nlane, key1 = kt0 + 16 + nlane;
    float alpha[8];
#pragma unroll
    for (int r = 0; r < 8; ++r) {
      int   qrow = qt0 + r + 8 * half;
      float v0 = s0[r] * scale, v1 = s1[r] * scale;
      if (key0 > qrow) v0 = -3.0e38f;
      if (key1 > qrow) v1 = -3.0e38f;
      // row max across the 16-lane group holding this row
      float tm = fmaxf(v0, v1);
#pragma unroll
      for (int off = 1; off < 16; off <<= 1) tm = fmaxf(tm, __shfl_xor(tm, off, 32));
      float nm = fmaxf(mrow[r], tm);
      float al = __expf(mrow[r] - nm);
      float p0 = __expf(v0 - nm), p1 = __expf(v1 - nm);
      float rs = p0 + p1;
#pragma unroll
      for (int off = 1; off < 16; off <<= 1) rs += __shfl_xor(rs, off, 32);
      lrow[r]  = lrow[r] * al + rs;
      mrow[r]  = nm;
      alpha[r] = al;
      s0[r] = p0; s1[r] = p1;
    }

    // rescale running output accumulators
#pragma unroll
    for (int di = 0; di < 4; ++di)
#pragma unroll
      for (int r = 0; r < 8; ++r) oacc[di][r] *= alpha[r];

    // stage P (16x32) to LDS in row-major, reload as A fragment
#pragma unroll
    for (int r = 0; r < 8; ++r) {
      int row = r + 8 * half;
      sP[row * 32 + nlane]      = (__bf16)s0[r];
      sP[row * 32 + 16 + nlane] = (__bf16)s1[r];
    }
    __syncthreads();
    v16bf ap = load_a_frag(sP, 32, 0, 0, lane);

    // P (16x32) @ V (32 keys x 64 dh)  via transposed-V B fragments
#pragma unroll
    for (int di = 0; di < 4; ++di) {
      v16bf bv = load_b_frag(Vh, TT, di * 16, kt0, lane);
      oacc[di] = wmma_bf16(ap, bv, oacc[di]);
    }
    __syncthreads();
  }

  // ---- finalize: divide by l, store O in (B*T, D) bf16 row-major ----
  int b = bh >> 4, h = bh & (NHEADS - 1);
#pragma unroll
  for (int r = 0; r < 8; ++r) {
    float inv = 1.0f / lrow[r];
    int   t   = qt0 + r + 8 * half;
    size_t base = ((size_t)(b * TT + t)) * DMODEL + h * DHEAD;
#pragma unroll
    for (int di = 0; di < 4; ++di)
      O[base + di * 16 + nlane] = (__bf16)(oacc[di][r] * inv);
  }
}

// ---------------------------------------------------------------------------
// Kernel 6: output projection  out = O @ oW^T + ob   (f32 result)
// Same tiling as pid_gemm, single component, bias epilogue.
// ---------------------------------------------------------------------------
__global__ __launch_bounds__(256) void out_gemm_kernel(
    const __bf16* __restrict__ A, const __bf16* __restrict__ W,
    const float* __restrict__ bias, float* __restrict__ out) {
  int lane = threadIdx.x & 31;
  int wid  = threadIdx.x >> 5;
  int wm   = wid & 3, wn = wid >> 2;
  int m0   = blockIdx.x * 128 + wm * 32;
  int n0   = blockIdx.y * 64 + wn * 32;

  v8f acc[2][2] = {};
  for (int k0 = 0; k0 < DMODEL; k0 += 32) {
    v16bf a0 = load_a_frag(A, DMODEL, m0,      k0, lane);
    v16bf a1 = load_a_frag(A, DMODEL, m0 + 16, k0, lane);
    v16bf b0 = load_b_frag(W, DMODEL, n0,      k0, lane);
    v16bf b1 = load_b_frag(W, DMODEL, n0 + 16, k0, lane);
    acc[0][0] = wmma_bf16(a0, b0, acc[0][0]);
    acc[0][1] = wmma_bf16(a0, b1, acc[0][1]);
    acc[1][0] = wmma_bf16(a1, b0, acc[1][0]);
    acc[1][1] = wmma_bf16(a1, b1, acc[1][1]);
  }

  int half  = lane >> 4;
  int nlane = lane & 15;
#pragma unroll
  for (int mi = 0; mi < 2; ++mi)
#pragma unroll
    for (int ni = 0; ni < 2; ++ni) {
      int n = n0 + ni * 16 + nlane;
      float bv = bias[n];
#pragma unroll
      for (int r = 0; r < 8; ++r) {
        int m = m0 + mi * 16 + r + 8 * half;
        out[(size_t)m * DMODEL + n] = acc[mi][ni][r] + bv;
      }
    }
}

// ---------------------------------------------------------------------------
// Host launcher
// ---------------------------------------------------------------------------
extern "C" void kernel_launch(void* const* d_in, const int* in_sizes, int n_in,
                              void* d_out, int out_size, void* d_ws, size_t ws_size,
                              hipStream_t stream) {
  (void)in_sizes; (void)n_in; (void)out_size; (void)ws_size;

  // Input order: x, qWp,qWi,qWd,qWg,qbg, kWp,kWi,kWd,kWg,kbg,
  //              vWp,vWi,vWd,vWg,vbg, oW, ob
  const float* x   = (const float*)d_in[0];
  const float* qWg = (const float*)d_in[4];
  const float* qbg = (const float*)d_in[5];
  const float* kWg = (const float*)d_in[9];
  const float* kbg = (const float*)d_in[10];
  const float* vWg = (const float*)d_in[14];
  const float* vbg = (const float*)d_in[15];
  const float* ob  = (const float*)d_in[17];
  float* out = (float*)d_out;

  // Workspace layout
  const size_t MTD = (size_t)MTOK * DMODEL;      // 4M elements
  const size_t DD  = (size_t)DMODEL * DMODEL;    // 1M elements
  __bf16* xbf  = (__bf16*)d_ws;
  __bf16* intb = xbf  + MTD;
  __bf16* derb = intb + MTD;
  __bf16* Qb   = derb + MTD;
  __bf16* Kb   = Qb   + MTD;
  __bf16* Vtb  = Kb   + MTD;
  __bf16* Ob   = Vtb  + MTD;
  __bf16* Wb   = Ob   + MTD;                     // 10 * DD bf16
  float*  g    = (float*)(Wb + 10 * DD);         // 3 * MTOK * 3 f32

  // 1) weights -> bf16
  WPtrs wp;
  wp.p[0] = (const float*)d_in[1];  wp.p[1] = (const float*)d_in[2];
  wp.p[2] = (const float*)d_in[3];  wp.p[3] = (const float*)d_in[6];
  wp.p[4] = (const float*)d_in[7];  wp.p[5] = (const float*)d_in[8];
  wp.p[6] = (const float*)d_in[11]; wp.p[7] = (const float*)d_in[12];
  wp.p[8] = (const float*)d_in[13]; wp.p[9] = (const float*)d_in[16];
  convert_weights_kernel<<<(10 * DD) / 256, 256, 0, stream>>>(wp, Wb);

  // 2) gates
  pid_gates_kernel<<<(3 * MTOK) / 4, 128, 0, stream>>>(
      x, qWg, qbg, kWg, kbg, vWg, vbg, g);

  // 3) scan (integral / derivative) + bf16 conversion of x
  pid_scan_kernel<<<(BB * DMODEL) / 256, 256, 0, stream>>>(x, xbf, intb, derb);

  // 4) the three PID projection GEMMs
  dim3 ggrid(MTOK / 128, DMODEL / 64);
  for (int p = 0; p < 3; ++p) {
    __bf16* dst = (p == 0) ? Qb : (p == 1) ? Kb : Vtb;
    pid_gemm_kernel<<<ggrid, 256, 0, stream>>>(
        xbf, intb, derb,
        Wb + (size_t)(p * 3 + 0) * DD,
        Wb + (size_t)(p * 3 + 1) * DD,
        Wb + (size_t)(p * 3 + 2) * DD,
        g + (size_t)p * MTOK * 3,
        dst, p);
  }

  // 5) causal flash attention
  flash_attn_kernel<<<dim3(BB * NHEADS, TT / 16), 32, 0, stream>>>(Qb, Kb, Vtb, Ob);

  // 6) output projection + bias -> f32 d_out
  out_gemm_kernel<<<ggrid, 256, 0, stream>>>(Ob, Wb + 9 * DD, ob, out);
}
